// GenePerceiverJEPA_88029649699164
// MI455X (gfx1250) — compile-verified, matrix operand order checked by reading, not run on hardware
//
#include <hip/hip_runtime.h>
#include <hip/hip_bf16.h>

// ---------------------------------------------------------------------------
// GenePerceiverJEPA forward for MI455X (gfx1250, wave32, WMMA).
// All matmuls -> v_wmma_f32_16x16x32_bf16 (bf16 in, fp32 accumulate).
// GEMM stages operands with global_load_async_to_lds_b128 (ASYNCcnt) into
// double-buffered LDS; flash-attention (online softmax) for both attentions.
// ---------------------------------------------------------------------------

typedef __bf16 bf16_t;
typedef __attribute__((ext_vector_type(16))) __bf16 v16bf;
typedef __attribute__((ext_vector_type(8)))  __bf16 v8bf;
typedef __attribute__((ext_vector_type(8)))  float  v8f;

#define CB  8      // batch
#define CN  4096   // tokens
#define CD  768    // d_model
#define CH  12     // heads
#define CL  512    // latents
#define CDBLK 6
#define CDH 64
#define CFF 3072

// ---- WMMA helpers ---------------------------------------------------------

__device__ __forceinline__ v8f wmma_bf16(v16bf a, v16bf b, v8f c) {
  // 8 args: (neg_a, A, neg_b, B, c_mod, C, reuse_a, reuse_b)
  return __builtin_amdgcn_wmma_f32_16x16x32_bf16(false, a, false, b, (short)0, c,
                                                 false, false);
}

// Load one 16x32 bf16 A/B fragment row for this lane.
// ISA layout (16-bit A 16x32): lanes 0-15 hold K=0..7 & 16..23,
// lanes 16-31 hold K=8..15 & 24..31 (pairs packed per VGPR).
// `base` points at element K=0 of this lane's matrix row; khalf = lane>>4.
__device__ __forceinline__ v16bf ld_frag(const bf16_t* base, int khalf) {
  union { v16bf v; v8bf h[2]; } u;
  u.h[0] = *(const v8bf*)(base + (khalf << 3));
  u.h[1] = *(const v8bf*)(base + 16 + (khalf << 3));
  return u.v;
}

// Async global->LDS 16B copy (per lane), tracked with ASYNCcnt.
__device__ __forceinline__ void async_copy16(void* lds_ptr, const void* gptr) {
#if defined(__gfx1250__)
  unsigned loff = (unsigned)(unsigned long long)(uintptr_t)lds_ptr;
  unsigned long long ga = (unsigned long long)(uintptr_t)gptr;
  asm volatile("global_load_async_to_lds_b128 %0, %1, off"
               :: "v"(loff), "v"(ga) : "memory");
#else
  *(v8bf*)lds_ptr = *(const v8bf*)gptr;
#endif
}

__device__ __forceinline__ void async_wait0() {
#if defined(__gfx1250__)
  asm volatile("s_wait_asynccnt 0" ::: "memory");
#endif
}

__device__ __forceinline__ float gelu_exact(float v) {
  return 0.5f * v * (1.0f + erff(v * 0.70710678118654752f));
}

// ---- fp32 -> bf16 convert -------------------------------------------------

__global__ void cvt_bf16_kernel(const float* __restrict__ in,
                                bf16_t* __restrict__ out, long long n) {
  long long i = (long long)blockIdx.x * blockDim.x + threadIdx.x;
  long long stride = (long long)gridDim.x * blockDim.x;
  for (; i < n; i += stride) out[i] = (bf16_t)in[i];
}

// ---- LayerNorm over D=768, one block per row ------------------------------

__global__ __launch_bounds__(256) void ln_kernel(const float* __restrict__ X,
                                                 const float* __restrict__ g,
                                                 const float* __restrict__ bta,
                                                 float* __restrict__ outf,
                                                 bf16_t* __restrict__ outb) {
  int row = blockIdx.x;
  int tid = threadIdx.x;
  const float* x = X + (size_t)row * CD;
  __shared__ float red[256];

  float s = 0.f;
  for (int i = tid; i < CD; i += 256) s += x[i];
  red[tid] = s;
  __syncthreads();
  for (int off = 128; off > 0; off >>= 1) {
    if (tid < off) red[tid] += red[tid + off];
    __syncthreads();
  }
  float mu = red[0] * (1.0f / CD);
  __syncthreads();

  float v = 0.f;
  for (int i = tid; i < CD; i += 256) { float d = x[i] - mu; v += d * d; }
  red[tid] = v;
  __syncthreads();
  for (int off = 128; off > 0; off >>= 1) {
    if (tid < off) red[tid] += red[tid + off];
    __syncthreads();
  }
  float inv = rsqrtf(red[0] * (1.0f / CD) + 1e-5f);

  for (int i = tid; i < CD; i += 256) {
    float y = (x[i] - mu) * inv * g[i] + bta[i];
    if (outf) outf[(size_t)row * CD + i] = y;
    if (outb) outb[(size_t)row * CD + i] = (bf16_t)y;
  }
}

// ---- Generic WMMA GEMM: C[M,Nout] = act(A[M,K] @ W[Nout,K]^T + bias) (+res)
// Block: 256 thr (8 waves), 64x128 output tile. K staged 32-wide through
// double-buffered LDS with async copies; waves 4x2, each 16 rows x 64 cols
// -> 4 WMMAs per K-step, one barrier per K-step.

template <int OUT_BF16, int ACT_GELU>
__global__ __launch_bounds__(256) void gemm_kernel(
    const bf16_t* __restrict__ A, const bf16_t* __restrict__ W,
    const float* __restrict__ bias, const float* __restrict__ residual,
    int res_mod, float* __restrict__ Cf, bf16_t* __restrict__ Cb,
    int M, int K, int Nout) {
  __shared__ bf16_t sA[2][64 * 32];
  __shared__ bf16_t sB[2][128 * 32];

  int tid = threadIdx.x;
  int lane = tid & 31, wave = tid >> 5;
  int wm = wave & 3, wn = wave >> 2;             // 4x2 wave grid
  int bm = blockIdx.y * 64, bn = blockIdx.x * 128;
  int khalf = lane >> 4, lrow = lane & 15;

  v8f acc0, acc1, acc2, acc3;
  for (int i = 0; i < 8; ++i) { acc0[i] = 0.f; acc1[i] = 0.f; acc2[i] = 0.f; acc3[i] = 0.f; }

  // staging assignment: A 64x32 (1x16B/thread), B 128x32 (2x16B/thread)
  int lrA = tid >> 2, lcA = (tid & 3) * 8;
  int lrB = tid >> 1, lcB = (tid & 1) * 16;
  const bf16_t* aptr = A + (size_t)(bm + lrA) * K + lcA;
  const bf16_t* wptr = W + (size_t)(bn + lrB) * K + lcB;
  bf16_t* aSlot[2] = { &sA[0][lrA * 32 + lcA], &sA[1][lrA * 32 + lcA] };
  bf16_t* bSlot[2] = { &sB[0][lrB * 32 + lcB], &sB[1][lrB * 32 + lcB] };

  // prologue: stage K-tile 0 into buffer 0
  async_copy16(aSlot[0], aptr);
  async_copy16(bSlot[0], wptr);
  async_copy16(bSlot[0] + 8, wptr + 8);
  async_wait0();
  __syncthreads();

  int nk = K / 32;
  for (int it = 0; it < nk; ++it) {
    int cur = it & 1;
    if (it + 1 < nk) {  // issue async copies for the next K-tile
      int nxt = cur ^ 1;
      const bf16_t* an = aptr + (size_t)(it + 1) * 32;
      const bf16_t* wn_ = wptr + (size_t)(it + 1) * 32;
      async_copy16(aSlot[nxt], an);
      async_copy16(bSlot[nxt], wn_);
      async_copy16(bSlot[nxt] + 8, wn_ + 8);
    }
    const bf16_t* pa = &sA[cur][(wm * 16 + lrow) * 32];
    const bf16_t* pb = &sB[cur][(wn * 64 + lrow) * 32];
    v16bf af = ld_frag(pa, khalf);
    v16bf b0 = ld_frag(pb, khalf);
    v16bf b1 = ld_frag(pb + 16 * 32, khalf);
    v16bf b2 = ld_frag(pb + 32 * 32, khalf);
    v16bf b3 = ld_frag(pb + 48 * 32, khalf);
    acc0 = wmma_bf16(af, b0, acc0);
    acc1 = wmma_bf16(af, b1, acc1);
    acc2 = wmma_bf16(af, b2, acc2);
    acc3 = wmma_bf16(af, b3, acc3);
    if (it + 1 < nk) async_wait0();
    __syncthreads();
  }

  // C fragment: lane<16 -> rows 0..7 (vgpr r), col=lane; lane>=16 -> rows 8..15
  int r0 = wm * 16 + khalf * 8;
  int c0 = bn + wn * 64 + lrow;
  for (int r = 0; r < 8; ++r) {
    int row = bm + r0 + r;
    float v0 = acc0[r], v1 = acc1[r], v2 = acc2[r], v3 = acc3[r];
    if (bias) {
      v0 += bias[c0]; v1 += bias[c0 + 16]; v2 += bias[c0 + 32]; v3 += bias[c0 + 48];
    }
    if (ACT_GELU) {
      v0 = gelu_exact(v0); v1 = gelu_exact(v1);
      v2 = gelu_exact(v2); v3 = gelu_exact(v3);
    }
    if (residual) {
      size_t rr = res_mod ? (size_t)(row % res_mod) : (size_t)row;
      const float* res = residual + rr * Nout;
      v0 += res[c0]; v1 += res[c0 + 16]; v2 += res[c0 + 32]; v3 += res[c0 + 48];
    }
    if (OUT_BF16) {
      bf16_t* crow = Cb + (size_t)row * Nout;
      crow[c0] = (bf16_t)v0; crow[c0 + 16] = (bf16_t)v1;
      crow[c0 + 32] = (bf16_t)v2; crow[c0 + 48] = (bf16_t)v3;
    } else {
      float* crow = Cf + (size_t)row * Nout;
      crow[c0] = v0; crow[c0 + 16] = v1; crow[c0 + 32] = v2; crow[c0 + 48] = v3;
    }
  }
}

// ---- V transpose: [B,NK,row_stride] slice -> VT[B,H,DH,NK] ----------------

__global__ void transpose_v_kernel(const bf16_t* __restrict__ src,
                                   size_t row_stride, int col_off,
                                   bf16_t* __restrict__ dst, int NK,
                                   long long total) {
  long long idx = (long long)blockIdx.x * blockDim.x + threadIdx.x;
  if (idx >= total) return;
  int per_bh = NK * CDH;
  int bh = (int)(idx / per_bh);
  int rem = (int)(idx % per_bh);
  int n = rem >> 6;
  int dh = rem & 63;
  int b = bh / CH, h = bh % CH;
  dst[((size_t)bh * CDH + dh) * NK + n] =
      src[((size_t)b * NK + n) * row_stride + col_off + h * CDH + dh];
}

// ---- Flash attention (online softmax), wave32 WMMA ------------------------
// Each wave owns a 16-row Q tile; 8 waves/block => 128 rows; grid = B*H*(Lq/128).
// Q/K read strided in place; V read from VT [B,H,DH,NK].
// K fragments loaded as a batch before the QK WMMAs; V fragments hoisted
// before the softmax barrier so global latency overlaps the exp stage.

__global__ __launch_bounds__(256) void flash_kernel(
    const bf16_t* __restrict__ Q, size_t qB, size_t qR,
    const bf16_t* __restrict__ Kp, size_t kB, size_t kR,
    const bf16_t* __restrict__ VT, const int* __restrict__ lengths,
    bf16_t* __restrict__ Out, int Hn, int Lq, int NK, float scale) {
  __shared__ float  sS[8][16][32];
  __shared__ bf16_t sP[8][16][32];
  __shared__ float  sM[8][16];
  __shared__ float  sL[8][16];
  __shared__ float  sAl[8][16];

  int tid = threadIdx.x, lane = tid & 31, wave = tid >> 5;
  int khalf = lane >> 4, lrow = lane & 15;

  int tiles = Lq / 128;
  int bh = blockIdx.x / tiles;
  int lt = blockIdx.x % tiles;
  int b = bh / Hn, h = bh % Hn;
  int rowbase = lt * 128 + wave * 16;

  const bf16_t* qbase = Q + (size_t)b * qB + (size_t)h * CDH +
                        (size_t)(rowbase + lrow) * qR;
  v16bf qf0 = ld_frag(qbase, khalf);        // K(dh) 0..31
  v16bf qf1 = ld_frag(qbase + 32, khalf);   // K(dh) 32..63
  const bf16_t* kbase = Kp + (size_t)b * kB + (size_t)h * CDH;
  const bf16_t* vbase = VT + (size_t)bh * CDH * (size_t)NK;

  v8f o0, o1, o2, o3;
  for (int i = 0; i < 8; ++i) { o0[i] = 0.f; o1[i] = 0.f; o2[i] = 0.f; o3[i] = 0.f; }

  if (lane < 16) { sM[wave][lane] = -1e30f; sL[wave][lane] = 0.f; }
  int len = lengths ? lengths[b] : NK;
  int nk_eff = (len + 31) & ~31;
  if (nk_eff > NK) nk_eff = NK;
  __syncthreads();

  for (int n0 = 0; n0 < nk_eff; n0 += 32) {
    // --- load all K fragments, then 4 QK WMMAs (K-dim = DH = 64)
    const bf16_t* krow0 = kbase + (size_t)(n0 + lrow) * kR;
    const bf16_t* krow1 = kbase + (size_t)(n0 + 16 + lrow) * kR;
    v16bf kf00 = ld_frag(krow0, khalf);
    v16bf kf01 = ld_frag(krow0 + 32, khalf);
    v16bf kf10 = ld_frag(krow1, khalf);
    v16bf kf11 = ld_frag(krow1 + 32, khalf);
    if (n0 + 32 < nk_eff)
      __builtin_prefetch(kbase + (size_t)(n0 + 32 + lrow) * kR, 0, 0);
    v8f s0, s1;
    for (int i = 0; i < 8; ++i) { s0[i] = 0.f; s1[i] = 0.f; }
    s0 = wmma_bf16(qf0, kf00, s0);
    s0 = wmma_bf16(qf1, kf01, s0);
    s1 = wmma_bf16(qf0, kf10, s1);
    s1 = wmma_bf16(qf1, kf11, s1);
    for (int r = 0; r < 8; ++r) {
      sS[wave][khalf * 8 + r][lrow]      = s0[r];
      sS[wave][khalf * 8 + r][16 + lrow] = s1[r];
    }
    // --- hoist V fragments (consumed after softmax)
    v16bf vf0 = ld_frag(vbase + (size_t)(lrow) * NK + n0, khalf);
    v16bf vf1 = ld_frag(vbase + (size_t)(16 + lrow) * NK + n0, khalf);
    v16bf vf2 = ld_frag(vbase + (size_t)(32 + lrow) * NK + n0, khalf);
    v16bf vf3 = ld_frag(vbase + (size_t)(48 + lrow) * NK + n0, khalf);
    __syncthreads();

    // --- online softmax: lanes 0..15 each own one of the wave's 16 rows
    if (lane < 16) {
      int r = lane;
      float mold = sM[wave][r];
      int nv = len - n0; if (nv > 32) nv = 32; if (nv < 0) nv = 0;
      float mx = mold;
      for (int c = 0; c < nv; ++c) {
        float sv = sS[wave][r][c] * scale;
        sS[wave][r][c] = sv;
        mx = fmaxf(mx, sv);
      }
      float alpha = 1.0f, lsum = 0.0f;
      if (mx > -1e29f) {
        alpha = __expf(mold - mx);   // exp(-1e30) underflows to 0 on fresh rows
        for (int c = 0; c < 32; ++c) {
          float e = (c < nv) ? __expf(sS[wave][r][c] - mx) : 0.0f;
          sP[wave][r][c] = (bf16_t)e;
          lsum += e;
        }
      } else {
        for (int c = 0; c < 32; ++c) sP[wave][r][c] = (bf16_t)0.0f;
      }
      sL[wave][r] = sL[wave][r] * alpha + lsum;
      sM[wave][r] = mx;
      sAl[wave][r] = alpha;
    }
    __syncthreads();

    // --- rescale accumulators, then O += P @ V (K-dim = 32 keys)
    for (int r = 0; r < 8; ++r) {
      float a = sAl[wave][khalf * 8 + r];
      o0[r] *= a; o1[r] *= a; o2[r] *= a; o3[r] *= a;
    }
    v16bf pf = ld_frag(&sP[wave][lrow][0], khalf);
    o0 = wmma_bf16(pf, vf0, o0);
    o1 = wmma_bf16(pf, vf1, o1);
    o2 = wmma_bf16(pf, vf2, o2);
    o3 = wmma_bf16(pf, vf3, o3);
    __syncthreads();
  }

  // --- epilogue: O / (l + 1e-6), write merged-head bf16
  for (int r = 0; r < 8; ++r) {
    int rr = khalf * 8 + r;
    float linv = 1.0f / (sL[wave][rr] + 1e-6f);
    int row = rowbase + rr;
    bf16_t* orow = Out + ((size_t)b * Lq + row) * (size_t)(Hn * CDH) + h * CDH;
    orow[lrow]      = (bf16_t)(o0[r] * linv);
    orow[16 + lrow] = (bf16_t)(o1[r] * linv);
    orow[32 + lrow] = (bf16_t)(o2[r] * linv);
    orow[48 + lrow] = (bf16_t)(o3[r] * linv);
  }
}

// ---- mean over L per batch ------------------------------------------------

__global__ void mean_kernel(const float* __restrict__ xnf, float* __restrict__ out) {
  int b = blockIdx.x, tid = threadIdx.x;
  for (int d = tid; d < CD; d += 256) {
    float s = 0.f;
    const float* p = xnf + (size_t)b * CL * CD + d;
    for (int l = 0; l < CL; ++l) s += p[(size_t)l * CD];
    out[(size_t)b * CD + d] = s * (1.0f / CL);
  }
}

// ---------------------------------------------------------------------------

extern "C" void kernel_launch(void* const* d_in, const int* in_sizes, int n_in,
                              void* d_out, int out_size, void* d_ws, size_t ws_size,
                              hipStream_t stream) {
  (void)in_sizes; (void)n_in; (void)out_size; (void)ws_size;
  const float* tokens  = (const float*)d_in[0];
  const int*   lengths = (const int*)d_in[1];
  const float* latents = (const float*)d_in[2];
  const float* lnq_g   = (const float*)d_in[3];
  const float* lnq_b   = (const float*)d_in[4];
  const float* lnkv_g  = (const float*)d_in[5];
  const float* lnkv_b  = (const float*)d_in[6];
  const float* Wq      = (const float*)d_in[7];
  const float* Wkv     = (const float*)d_in[8];
  const float* Wo      = (const float*)d_in[9];
  const float* bo      = (const float*)d_in[10];
  const float* ln1g    = (const float*)d_in[11];
  const float* ln1b    = (const float*)d_in[12];
  const float* inw     = (const float*)d_in[13];
  const float* inb     = (const float*)d_in[14];
  const float* outw    = (const float*)d_in[15];
  const float* outb    = (const float*)d_in[16];
  const float* ln2g    = (const float*)d_in[17];
  const float* ln2b    = (const float*)d_in[18];
  const float* w1      = (const float*)d_in[19];
  const float* b1      = (const float*)d_in[20];
  const float* w2      = (const float*)d_in[21];
  const float* b2      = (const float*)d_in[22];
  const float* fng     = (const float*)d_in[23];
  const float* fnb     = (const float*)d_in[24];
  float* out = (float*)d_out;

  // --- workspace bump allocator (peak ~270 MB with phase overlays) ---------
  char* ws = (char*)d_ws;
  size_t off = 0;
  auto alloc_bf = [&](size_t elems) -> bf16_t* {
    bf16_t* p = (bf16_t*)(ws + off);
    off = (off + elems * sizeof(bf16_t) + 255) & ~(size_t)255;
    return p;
  };
  auto alloc_f = [&](size_t elems) -> float* {
    float* p = (float*)(ws + off);
    off = (off + elems * sizeof(float) + 255) & ~(size_t)255;
    return p;
  };

  bf16_t* wq_b   = alloc_bf((size_t)CD * CD);
  bf16_t* wkv_b  = alloc_bf((size_t)2 * CD * CD);
  bf16_t* wo_b   = alloc_bf((size_t)CD * CD);
  bf16_t* inw_b  = alloc_bf((size_t)CDBLK * 3 * CD * CD);
  bf16_t* outw_b = alloc_bf((size_t)CDBLK * CD * CD);
  bf16_t* w1_b   = alloc_bf((size_t)CDBLK * CFF * CD);
  bf16_t* w2_b   = alloc_bf((size_t)CDBLK * CD * CFF);
  float*  queries_f = alloc_f((size_t)CL * CD);
  bf16_t* queries_b = alloc_bf((size_t)CL * CD);
  bf16_t* qflat     = alloc_bf((size_t)CL * CD);
  float*  x_f       = alloc_f((size_t)CB * CL * CD);
  bf16_t* xn_b      = alloc_bf((size_t)CB * CL * CD);
  bf16_t* attn_b    = alloc_bf((size_t)CB * CL * CD);
  // transient region: tokens_norm dies after kv GEMM (vT overlays it, same size);
  // kvflat dies after cross-attn (block-phase transients overlay it).
  bf16_t* tokens_norm = alloc_bf((size_t)CB * CN * CD);
  bf16_t* kvflat      = alloc_bf((size_t)CB * CN * 2 * CD);
  bf16_t* vT  = tokens_norm;                                  // [B,H,DH,N]
  bf16_t* qkvflat = kvflat;                                   // [B*L, 3D]
  bf16_t* v2T = qkvflat + (size_t)CB * CL * 3 * CD;           // [B,H,DH,L]
  bf16_t* ffh = v2T + (size_t)CB * CH * CDH * CL;             // [B*L, FF]
  float*  xnf = (float*)(ffh + (size_t)CB * CL * CFF);        // [B*L, D]

  auto cvt = [&](const float* src, bf16_t* dst, long long n) {
    int blocks = (int)((n + 255) / 256);
    if (blocks > 16384) blocks = 16384;
    cvt_bf16_kernel<<<blocks, 256, 0, stream>>>(src, dst, n);
  };

  // --- weights to bf16 -----------------------------------------------------
  cvt(Wq,   wq_b,   (long long)CD * CD);
  cvt(Wkv,  wkv_b,  (long long)2 * CD * CD);
  cvt(Wo,   wo_b,   (long long)CD * CD);
  cvt(inw,  inw_b,  (long long)CDBLK * 3 * CD * CD);
  cvt(outw, outw_b, (long long)CDBLK * CD * CD);
  cvt(w1,   w1_b,   (long long)CDBLK * CFF * CD);
  cvt(w2,   w2_b,   (long long)CDBLK * CD * CFF);

  // --- LayerNorms (queries shared across batch) ----------------------------
  ln_kernel<<<CL, 256, 0, stream>>>(latents, lnq_g, lnq_b, queries_f, queries_b);
  ln_kernel<<<CB * CN, 256, 0, stream>>>(tokens, lnkv_g, lnkv_b, nullptr, tokens_norm);

  // --- q = LN(latents) @ Wq^T ; kv = LN(tokens) @ Wkv^T --------------------
  gemm_kernel<1, 0><<<dim3(CD / 128, CL / 64), 256, 0, stream>>>(
      queries_b, wq_b, nullptr, nullptr, 0, nullptr, qflat, CL, CD, CD);
  gemm_kernel<1, 0><<<dim3(2 * CD / 128, (CB * CN) / 64), 256, 0, stream>>>(
      tokens_norm, wkv_b, nullptr, nullptr, 0, nullptr, kvflat, CB * CN, CD, 2 * CD);

  // --- V transpose, masked chunked cross-attention -------------------------
  {
    long long tot = (long long)CB * CH * CDH * CN;
    transpose_v_kernel<<<(int)(tot / 256), 256, 0, stream>>>(
        kvflat, (size_t)2 * CD, CD, vT, CN, tot);
  }
  flash_kernel<<<CB * CH * (CL / 128), 256, 0, stream>>>(
      qflat, (size_t)0, (size_t)CD,                      // Q [L,768], shared over b
      kvflat, (size_t)CN * 2 * CD, (size_t)2 * CD,       // K in place (cols 0..767)
      vT, lengths, attn_b, CH, CL, CN, 0.125f);

  // --- x = queries + attn @ Wo^T + bo --------------------------------------
  gemm_kernel<0, 0><<<dim3(CD / 128, (CB * CL) / 64), 256, 0, stream>>>(
      attn_b, wo_b, bo, queries_f, CL, x_f, nullptr, CB * CL, CD, CD);

  // --- latent self-attention blocks ----------------------------------------
  for (int i = 0; i < CDBLK; ++i) {
    ln_kernel<<<CB * CL, 256, 0, stream>>>(x_f, ln1g + i * CD, ln1b + i * CD,
                                           nullptr, xn_b);
    gemm_kernel<1, 0><<<dim3(3 * CD / 128, (CB * CL) / 64), 256, 0, stream>>>(
        xn_b, inw_b + (size_t)i * 3 * CD * CD, inb + i * 3 * CD, nullptr, 0,
        nullptr, qkvflat, CB * CL, CD, 3 * CD);
    {
      long long tot = (long long)CB * CH * CDH * CL;
      transpose_v_kernel<<<(int)(tot / 256), 256, 0, stream>>>(
          qkvflat, (size_t)3 * CD, 2 * CD, v2T, CL, tot);
    }
    flash_kernel<<<CB * CH * (CL / 128), 256, 0, stream>>>(
        qkvflat, (size_t)CL * 3 * CD, (size_t)3 * CD,          // Q (cols 0..767)
        qkvflat + CD, (size_t)CL * 3 * CD, (size_t)3 * CD,     // K (cols 768..1535)
        v2T, nullptr, attn_b, CH, CL, CL, 0.125f);
    gemm_kernel<0, 0><<<dim3(CD / 128, (CB * CL) / 64), 256, 0, stream>>>(
        attn_b, outw_b + (size_t)i * CD * CD, outb + i * CD, x_f, 0,
        x_f, nullptr, CB * CL, CD, CD);
    ln_kernel<<<CB * CL, 256, 0, stream>>>(x_f, ln2g + i * CD, ln2b + i * CD,
                                           nullptr, xn_b);
    gemm_kernel<1, 1><<<dim3(CFF / 128, (CB * CL) / 64), 256, 0, stream>>>(
        xn_b, w1_b + (size_t)i * CFF * CD, b1 + i * CFF, nullptr, 0,
        nullptr, ffh, CB * CL, CD, CFF);
    gemm_kernel<0, 0><<<dim3(CD / 128, (CB * CL) / 64), 256, 0, stream>>>(
        ffh, w2_b + (size_t)i * CD * CFF, b2 + i * CD, x_f, 0,
        x_f, nullptr, CB * CL, CFF, CD);
  }

  // --- final LN + mean over latents ----------------------------------------
  ln_kernel<<<CB * CL, 256, 0, stream>>>(x_f, fng, fnb, xnf, nullptr);
  mean_kernel<<<CB, 256, 0, stream>>>(xnf, out);
}
